// BrushStroke_87368224735782
// MI455X (gfx1250) — matrix operand order repaired and checked by reading, not compiled, not run on hardware
//
#include <hip/hip_runtime.h>

// BrushStroke renderer for MI455X (gfx1250, wave32, WMMA).
// out[b,c,y,x] = (1/64) * sum_n  Fy_n (256x32) @ ( Fx_n (256x32) @ P_{n,c}^T (32x32) )^T
// Both GEMMs have K=32 -> one v_wmma_f32_16x16x32_f16 per 16x16 output tile.
//
// Pipeline notes:
//  - The t = Fx @ P^T staging through LDS is wave-private (wave w writes and
//    reads only rows [32w, 32w+32)), so the whole GEMM phase runs with NO
//    workgroup barriers; per-wave DS ordering + s_wait_dscnt is sufficient.
//  - Patches for iteration n+1 are prefetched into registers during the GEMM
//    phase of iteration n (hides global latency behind XDL work).

#define EPS    1e-7f
#define INV_S2 50.0f     // 1 / (2*0.1^2)

typedef __attribute__((ext_vector_type(16))) _Float16 v16h;
typedef __attribute__((ext_vector_type(8)))  float    v8f;

// ---------------- kernel 1: per-batch min/max normalization of brush coords --
__global__ void brush_norm_kernel(const float* __restrict__ brushes,
                                  float* __restrict__ gx, float* __restrict__ gy) {
  __shared__ float sx[64], sy[64];
  __shared__ float mnx, mxx, mny, mxy;
  const int b = blockIdx.x, n = threadIdx.x;
  const float x = brushes[(b * 64 + n) * 2 + 0];
  const float y = brushes[(b * 64 + n) * 2 + 1];
  sx[n] = x; sy[n] = y;
  __syncthreads();
  if (n == 0) {
    float mn = sx[0], mx = sx[0];
    for (int i = 1; i < 64; ++i) { mn = fminf(mn, sx[i]); mx = fmaxf(mx, sx[i]); }
    mnx = mn; mxx = mx;
    mn = sy[0]; mx = sy[0];
    for (int i = 1; i < 64; ++i) { mn = fminf(mn, sy[i]); mx = fmaxf(mx, sy[i]); }
    mny = mn; mxy = mx;
  }
  __syncthreads();
  gx[b * 64 + n] = (x - mnx) / (mxx - mnx + EPS) * 256.0f;
  gy[b * 64 + n] = (y - mny) / (mxy - mny + EPS) * 256.0f;
}

// K index for element e of a 16-bit A fragment (16x32 MxK), ISA layout:
// VGPR j<4: K = 2j,2j+1 (+8 for lanes 16-31); VGPR j>=4: K = 16+2(j-4)..(+8).
// Per lane this is two contiguous 8-half runs -> compiler emits ds_load_b128 x2.
__device__ __forceinline__ int a_frag_k(int e, int hi) {
  const int j = e >> 1;
  return ((j & 4) << 2) + hi * 8 + ((j & 3) << 1) + (e & 1);
}

// ---------------- kernel 2: main render ------------------------------------
__global__ __launch_bounds__(256) void brush_render_kernel(
    const float* __restrict__ patches, const float* __restrict__ gx,
    const float* __restrict__ gy, float* __restrict__ out) {
  __shared__ _Float16 sFx[256 * 32];   // Fx[x][q]  normalized, 16 KB
  __shared__ _Float16 sFy[16 * 32];    // Fy[ylocal][q]          1 KB
  __shared__ _Float16 sP[3 * 32 * 32]; // patch f16 [c][p][q]    6 KB
  __shared__ _Float16 sT[256 * 32];    // t[x][p] (wave-private rows), 16 KB
  __shared__ float sPart[2 * 8 * 32];  // partial denom sums     2 KB
  __shared__ float sInv[2 * 32];       // 1/(denom+eps)

  const int tid  = threadIdx.x;
  const int lane = tid & 31;
  const int wave = tid >> 5;
  const int hi   = (lane >> 4) & 1;
  const int lo   = lane & 15;
  const int b    = blockIdx.y;
  const int y0   = blockIdx.x * 16;

  const v8f vzero = {};
  v8f acc[3][2];
  for (int c = 0; c < 3; ++c)
    for (int i = 0; i < 2; ++i) acc[c][i] = vzero;

  // software pipeline: patch registers for the upcoming iteration
  float preg[12];
  {
    const float* pb = patches + (size_t)(b * 64 * 3) * 1024;
    for (int k = 0; k < 12; ++k) preg[k] = pb[tid + k * 256];
  }

  for (int n = 0; n < 64; ++n) {
    const float cgx = gx[b * 64 + n];
    const float cgy = gy[b * 64 + n];

    // -- stage A: partial normalization sums over the 288-wide padded domain
    {
      const int q = tid & 31, part = tid >> 5;          // 8 parts x 36 = 288
      const float ux = cgx + (float)q - 15.5f;
      const float uy = cgy + (float)q - 15.4f;
      float sxs = 0.f, sys = 0.f;
      for (int v = part * 36; v < part * 36 + 36; ++v) {
        const float a = (float)v - 16.0f;
        const float dx = a - ux; sxs += __expf(-dx * dx * INV_S2);
        const float dy = a - uy; sys += __expf(-dy * dy * INV_S2);
      }
      sPart[(0 * 8 + part) * 32 + q] = sxs;
      sPart[(1 * 8 + part) * 32 + q] = sys;
    }
    __syncthreads();
    // -- stage B: reduce to 1/(sum+eps)
    if (tid < 64) {
      const int xy = tid >> 5, q = tid & 31;
      float s = 0.f;
      for (int p = 0; p < 8; ++p) s += sPart[(xy * 8 + p) * 32 + q];
      sInv[xy * 32 + q] = 1.0f / (s + EPS);
    }
    __syncthreads();
    // -- stage C: build Fx (256x32), Fy (16x32), patch f16 from prefetched regs
    {
      const float xpos = (float)tid;                    // x = 0..255
      for (int q = 0; q < 32; ++q) {
        const float d = xpos - (cgx + (float)q - 15.5f);
        sFx[tid * 32 + q] = (_Float16)(__expf(-d * d * INV_S2) * sInv[q]);
      }
      for (int k = 0; k < 2; ++k) {                     // 512 Fy entries
        const int idx = tid + k * 256;
        const int yl = idx >> 5, q = idx & 31;
        const float d = (float)(y0 + yl) - (cgy + (float)q - 15.4f);
        sFy[idx] = (_Float16)(__expf(-d * d * INV_S2) * sInv[32 + q]);
      }
      for (int k = 0; k < 12; ++k)                      // 3072 patch values
        sP[tid + k * 256] = (_Float16)preg[k];
    }
    __syncthreads();

    // -- prefetch next iteration's patches; latency hidden behind GEMM phase
    if (n + 1 < 64) {
      const float* pb = patches + (size_t)((b * 64 + n + 1) * 3) * 1024;
      for (int k = 0; k < 12; ++k) preg[k] = pb[tid + k * 256];
    }

    // A fragment for GEMM2: Fy rows (M = y-local), K = p, shared across tiles
    v16h aFy;
    for (int e = 0; e < 16; ++e) aFy[e] = sFy[lo * 32 + a_frag_k(e, hi)];

    // ---- GEMM phase: barrier-free; all LDS traffic below is wave-private ---
    for (int c = 0; c < 3; ++c) {
      // GEMM1: t = Fx @ P_c^T ; wave handles x-tiles {2w, 2w+1}
      for (int i = 0; i < 2; ++i) {
        const int x0 = (wave * 2 + i) * 16;
        v16h aFx;  // rows [32w, 32w+32) were written by this wave in stage C
        for (int e = 0; e < 16; ++e) aFx[e] = sFx[(x0 + lo) * 32 + a_frag_k(e, hi)];
        for (int half = 0; half < 2; ++half) {          // p in [0,16) / [16,32)
          v16h bP;  // B[q, p] = P_c[p][q]; col = p, K = a_frag_k(e, hi)
          for (int e = 0; e < 16; ++e)
            bP[e] = sP[c * 1024 + (half * 16 + lo) * 32 + a_frag_k(e, hi)];
          v8f d = vzero;
          d = __builtin_amdgcn_wmma_f32_16x16x32_f16(
              false, aFx, false, bP, (short)0, d, false, false);
          for (int r = 0; r < 8; ++r)                   // D row = r + 8*hi
            sT[(x0 + r + 8 * hi) * 32 + half * 16 + lo] = (_Float16)d[r];
        }
      }
      // GEMM2: acc_c += Fy @ t^T; reads only this wave's own sT rows
      for (int i = 0; i < 2; ++i) {
        const int x0 = (wave * 2 + i) * 16;
        v16h bT;  // B[p, x] = t[x][p]; col = x, K = a_frag_k(e, hi)
        for (int e = 0; e < 16; ++e) bT[e] = sT[(x0 + lo) * 32 + a_frag_k(e, hi)];
        acc[c][i] = __builtin_amdgcn_wmma_f32_16x16x32_f16(
            false, aFy, false, bT, (short)0, acc[c][i], false, false);
      }
    }
  }

  // -- writeback: out[b][c][y][x] = acc / 64
  const float scale = 1.0f / 64.0f;
  for (int c = 0; c < 3; ++c)
    for (int i = 0; i < 2; ++i) {
      const int x = (wave * 2 + i) * 16 + lo;
      for (int r = 0; r < 8; ++r) {
        const int y = y0 + r + 8 * hi;
        out[(((size_t)b * 3 + c) * 256 + y) * 256 + x] = acc[c][i][r] * scale;
      }
    }
}

extern "C" void kernel_launch(void* const* d_in, const int* in_sizes, int n_in,
                              void* d_out, int out_size, void* d_ws, size_t ws_size,
                              hipStream_t stream) {
  const float* brushes = (const float*)d_in[0];   // (16, 64, 2) f32
  const float* patches = (const float*)d_in[1];   // (16, 64, 3, 32, 32) f32
  float* gx = (float*)d_ws;                       // 16*64 centers (x)
  float* gy = gx + 16 * 64;                       // 16*64 centers (y)

  brush_norm_kernel<<<16, 64, 0, stream>>>(brushes, gx, gy);

  dim3 grid(16, 16);                              // (y-tile, batch)
  brush_render_kernel<<<grid, 256, 0, stream>>>(patches, gx, gy, (float*)d_out);
}
